// MMDiTAttention_3418793967751
// MI455X (gfx1250) — matrix operand order repaired and checked by reading, not compiled, not run on hardware
//
#include <hip/hip_runtime.h>
#include <stdint.h>

typedef __attribute__((ext_vector_type(16))) __bf16 v16bf;
typedef __attribute__((ext_vector_type(8)))  float v8f;

union Frag16 { v16bf v; uint32_t d[8]; };

static __device__ __forceinline__ uint16_t f2bf(float f) {
    union { float f; uint32_t u; } v; v.f = f;
    uint32_t r = v.u + 0x7FFFu + ((v.u >> 16) & 1u);   // round-to-nearest-even
    return (uint16_t)(r >> 16);
}
static __device__ __forceinline__ uint32_t pack2bf(float a, float b) {
    return (uint32_t)f2bf(a) | ((uint32_t)f2bf(b) << 16);
}

// Minimal workgroup barrier: drain only own DS ops, then split-barrier.
// (Unlike __syncthreads(), does NOT wait outstanding global loads, so
// next-tile loads stay in flight across the barrier.)
static __device__ __forceinline__ void lds_barrier() {
    asm volatile(
        "s_wait_dscnt 0\n\t"
        "s_barrier_signal -1\n\t"
        "s_barrier_wait -1"
        ::: "memory");
}

// ---------------------------------------------------------------------------
// Elementwise fp32 -> bf16 (n multiple of 4).
// ---------------------------------------------------------------------------
__global__ __launch_bounds__(256) void convert_bf16_kernel(
    const float* __restrict__ X, uint16_t* __restrict__ Y, int n)
{
    const int i = (blockIdx.x * 256 + threadIdx.x) * 4;
    if (i < n) {
        const float4 v = *(const float4*)(X + i);
        uint2 o;
        o.x = pack2bf(v.x, v.y);
        o.y = pack2bf(v.z, v.w);
        *(uint2*)(Y + i) = o;
    }
}

// ---------------------------------------------------------------------------
// W [K,N] fp32 -> Wt [N,K] bf16 (pre-transposed). 64x64 tiles via LDS.
// Grid: (N/64, K/64).
// ---------------------------------------------------------------------------
__global__ __launch_bounds__(256) void transpose_w_kernel(
    const float* __restrict__ W, uint16_t* __restrict__ Wt, int K, int N)
{
    __shared__ uint16_t T[64][72];
    const int k0 = blockIdx.y * 64;
    const int n0 = blockIdx.x * 64;
    const int t  = threadIdx.x;

    {
        const int kr = t >> 2;             // 0..63
        const int nc = (t & 3) * 16;       // 0/16/32/48
        const float* wp = W + (size_t)(k0 + kr) * N + n0 + nc;
        #pragma unroll
        for (int i = 0; i < 16; ++i)
            T[nc + i][kr] = f2bf(wp[i]);
    }
    __syncthreads();
    {
        const int nr = t >> 2;
        const int kc = (t & 3) * 16;
        uint4* dst = (uint4*)(Wt + (size_t)(n0 + nr) * K + k0 + kc);
        const uint4* src = (const uint4*)&T[nr][kc];
        dst[0] = src[0];
        dst[1] = src[1];
    }
}

// ---------------------------------------------------------------------------
// GEMM: Y[M,N] = Xb[M,K] * Wt[N,K]^T + bias[N]  (bf16 in, fp32 out)
// Block 256 threads (8 wave32). Tile 128x128, BK=64, double-buffered LDS.
// lds_barrier() waits only dscnt, so next-tile global loads stay in flight
// across the barrier and their loadcnt wait lands AFTER the 16 WMMAs.
// ---------------------------------------------------------------------------
__global__ __launch_bounds__(256, 1) void gemm_bf16_kernel(
    const uint16_t* __restrict__ Xb, const uint16_t* __restrict__ Wt,
    const float* __restrict__ bias, float* __restrict__ Y,
    int M, int N, int K)
{
    __shared__ uint16_t Al[2][128][72];   // [buf][m][k]
    __shared__ uint16_t Bl[2][128][72];   // [buf][n][k]

    const int tid  = threadIdx.x;
    const int bm   = blockIdx.y * 128;
    const int bn   = blockIdx.x * 128;
    const int wv   = tid >> 5;
    const int lane = tid & 31;
    const int lh   = lane >> 4;
    const int l15  = lane & 15;
    const int wm   = (wv & 3) * 32;
    const int wn   = (wv >> 2) * 64;

    v8f acc[2][4] = {};

    const int sr = tid >> 1;           // staging row 0..127
    const int sc = (tid & 1) * 32;     // staging col 0 or 32

    const uint16_t* ap = Xb + (size_t)(bm + sr) * K + sc;
    const uint16_t* bp = Wt + (size_t)(bn + sr) * K + sc;

    // prologue: stage tile 0 into buffer 0 (batched loads, one wait)
    {
        uint4 ra0, ra1, ra2, ra3, rb0, rb1, rb2, rb3;
        const uint4* a4 = (const uint4*)ap;
        const uint4* b4 = (const uint4*)bp;
        ra0 = a4[0]; ra1 = a4[1]; ra2 = a4[2]; ra3 = a4[3];
        rb0 = b4[0]; rb1 = b4[1]; rb2 = b4[2]; rb3 = b4[3];
        uint4* da = (uint4*)&Al[0][sr][sc];
        uint4* db = (uint4*)&Bl[0][sr][sc];
        da[0] = ra0; da[1] = ra1; da[2] = ra2; da[3] = ra3;
        db[0] = rb0; db[1] = rb1; db[2] = rb2; db[3] = rb3;
    }

    int cur = 0;
    for (int kt = 0; kt < K; kt += 64) {
        const bool has_next = (kt + 64) < K;
        uint4 ra0, ra1, ra2, ra3, rb0, rb1, rb2, rb3;
        if (has_next) {
            const uint4* a4 = (const uint4*)(ap + kt + 64);
            const uint4* b4 = (const uint4*)(bp + kt + 64);
            ra0 = a4[0]; ra1 = a4[1]; ra2 = a4[2]; ra3 = a4[3];
            rb0 = b4[0]; rb1 = b4[1]; rb2 = b4[2]; rb3 = b4[3];
        }
        if (kt + 128 < K) {
            __builtin_prefetch(ap + kt + 128, 0, 1);
            __builtin_prefetch(bp + kt + 128, 0, 1);
        }
        lds_barrier();   // make buf[cur] stores visible; loads stay in flight

        #pragma unroll
        for (int kc = 0; kc < 2; ++kc) {
            Frag16 af[2];
            #pragma unroll
            for (int i = 0; i < 2; ++i) {
                const uint16_t* rp = &Al[cur][wm + i*16 + l15][kc*32 + lh*16];
                #pragma unroll
                for (int r = 0; r < 4; ++r) {
                    af[i].d[r]     = *(const uint32_t*)(rp + 2*r);
                    af[i].d[r + 4] = *(const uint32_t*)(rp + 8 + 2*r);
                }
            }
            #pragma unroll
            for (int j = 0; j < 4; ++j) {
                Frag16 bfr;
                const uint16_t* rp = &Bl[cur][wn + j*16 + l15][kc*32 + lh*16];
                #pragma unroll
                for (int r = 0; r < 8; ++r)
                    bfr.d[r] = *(const uint32_t*)(rp + 2*r);
                #pragma unroll
                for (int i = 0; i < 2; ++i)
                    acc[i][j] = __builtin_amdgcn_wmma_f32_16x16x32_bf16(
                        false, af[i].v, false, bfr.v, (short)0, acc[i][j], false, false);
            }
        }

        if (has_next) {   // loadcnt wait happens here, after the WMMAs
            uint4* da = (uint4*)&Al[cur ^ 1][sr][sc];
            uint4* db = (uint4*)&Bl[cur ^ 1][sr][sc];
            da[0] = ra0; da[1] = ra1; da[2] = ra2; da[3] = ra3;
            db[0] = rb0; db[1] = rb1; db[2] = rb2; db[3] = rb3;
        }
        cur ^= 1;
    }

    #pragma unroll
    for (int j = 0; j < 4; ++j) {
        const int col = bn + wn + j*16 + l15;
        const float bv = bias[col];
        #pragma unroll
        for (int i = 0; i < 2; ++i) {
            #pragma unroll
            for (int r = 0; r < 8; ++r) {
                const int row = bm + wm + i*16 + lh*8 + r;
                Y[(size_t)row * N + col] = acc[i][j][r] + bv;
            }
        }
    }
}

// ---------------------------------------------------------------------------
// RMSNorm(q,k) + RoPE(q,k) + pack q/k/v to bf16 [B,16,1792,64].
// The softmax scale 1/sqrt(64)=0.125 is folded into Q here (exact in bf16).
// ---------------------------------------------------------------------------
__global__ __launch_bounds__(256) void pack_norm_rope_kernel(
    const float* __restrict__ qkv, const float* __restrict__ qw,
    const float* __restrict__ kw,  const float* __restrict__ cosb,
    const float* __restrict__ sinb,
    uint16_t* __restrict__ qo, uint16_t* __restrict__ ko, uint16_t* __restrict__ vo,
    int Lp, int l_off)
{
    __shared__ float red[256];
    const int row = blockIdx.x;
    const int b   = row / Lp;
    const int ll  = row - b * Lp;
    const int l   = l_off + ll;
    const int tid = threadIdx.x;
    const int c0  = tid * 4;

    const float* src = qkv + (size_t)row * 3072;

    const float4 qv = *(const float4*)(src + c0);
    const float4 kv = *(const float4*)(src + 1024 + c0);
    const float4 vv = *(const float4*)(src + 2048 + c0);

    red[tid] = qv.x*qv.x + qv.y*qv.y + qv.z*qv.z + qv.w*qv.w;
    __syncthreads();
    for (int s = 128; s >= 1; s >>= 1) {
        if (tid < s) red[tid] += red[tid + s];
        __syncthreads();
    }
    const float rq = rsqrtf(red[0] * (1.0f / 1024.0f) + 1e-6f) * 0.125f; // + attn scale
    __syncthreads();
    red[tid] = kv.x*kv.x + kv.y*kv.y + kv.z*kv.z + kv.w*kv.w;
    __syncthreads();
    for (int s = 128; s >= 1; s >>= 1) {
        if (tid < s) red[tid] += red[tid + s];
        __syncthreads();
    }
    const float rk = rsqrtf(red[0] * (1.0f / 1024.0f) + 1e-6f);

    const int h  = c0 >> 6;
    const int dd = c0 & 63;
    const int j  = dd >> 1;
    const float c1 = cosb[l*32 + j],     s1 = sinb[l*32 + j];
    const float c2 = cosb[l*32 + j + 1], s2 = sinb[l*32 + j + 1];

    const size_t base = ((size_t)(b*16 + h) * 1792 + l) * 64 + dd;

    {
        const float x0 = qv.x * rq * qw[c0];
        const float x1 = qv.y * rq * qw[c0 + 1];
        const float x2 = qv.z * rq * qw[c0 + 2];
        const float x3 = qv.w * rq * qw[c0 + 3];
        uint2 o;
        o.x = pack2bf(x0*c1 - x1*s1, x0*s1 + x1*c1);
        o.y = pack2bf(x2*c2 - x3*s2, x2*s2 + x3*c2);
        *(uint2*)&qo[base] = o;
    }
    {
        const float x0 = kv.x * rk * kw[c0];
        const float x1 = kv.y * rk * kw[c0 + 1];
        const float x2 = kv.z * rk * kw[c0 + 2];
        const float x3 = kv.w * rk * kw[c0 + 3];
        uint2 o;
        o.x = pack2bf(x0*c1 - x1*s1, x0*s1 + x1*c1);
        o.y = pack2bf(x2*c2 - x3*s2, x2*s2 + x3*c2);
        *(uint2*)&ko[base] = o;
    }
    {
        uint2 o;
        o.x = pack2bf(vv.x, vv.y);
        o.y = pack2bf(vv.z, vv.w);
        *(uint2*)&vo[base] = o;
    }
}

// ---------------------------------------------------------------------------
// Flash attention. Block = 128 threads (4 waves); one (b,h), 64-row q tile;
// each wave owns 16 q rows. 28 key tiles of 64. Double-buffered K/V staging:
// one lds_barrier per tile, global loads overlap the WMMA/softmax work.
// ---------------------------------------------------------------------------
__global__ __launch_bounds__(128, 1) void attn_kernel(
    const uint16_t* __restrict__ qb, const uint16_t* __restrict__ kb,
    const uint16_t* __restrict__ vb, const int* __restrict__ mask,
    uint16_t* __restrict__ out_in, uint16_t* __restrict__ out_ctx)
{
    __shared__ uint16_t Kl[2][64][72];   // [buf][key][d]
    __shared__ uint16_t Vt[2][64][72];   // [buf][d][key]
    __shared__ uint16_t Pl[4][16][72];   // per-wave P tile [row][key]

    const int L = 1792;
    const int blk  = blockIdx.x;
    const int qt   = blk % 28;
    const int h    = (blk / 28) % 16;
    const int b    = blk / (28 * 16);
    const int tid  = threadIdx.x;
    const int wv   = tid >> 5;
    const int lane = tid & 31;
    const int lh   = lane >> 4;
    const int l15  = lane & 15;
    const int q0   = qt * 64 + wv * 16;

    const uint16_t* qbase = qb + (size_t)(b*16 + h) * L * 64;
    const uint16_t* kbase = kb + (size_t)(b*16 + h) * L * 64;
    const uint16_t* vbase = vb + (size_t)(b*16 + h) * L * 64;

    Frag16 qf[2];
    {
        const uint16_t* rp = qbase + (size_t)(q0 + l15) * 64 + lh * 16;
        #pragma unroll
        for (int kc = 0; kc < 2; ++kc) {
            const uint16_t* p = rp + kc * 32;
            #pragma unroll
            for (int r = 0; r < 4; ++r) {
                qf[kc].d[r]     = *(const uint32_t*)(p + 2*r);
                qf[kc].d[r + 4] = *(const uint32_t*)(p + 8 + 2*r);
            }
        }
    }

    v8f oacc[4] = {};
    float mrow[8], lrow[8];
    #pragma unroll
    for (int r = 0; r < 8; ++r) { mrow[r] = -3e38f; lrow[r] = 0.0f; }

    const int key = tid >> 1;
    const int d0  = (tid & 1) * 32;
    const uint2* gk0 = (const uint2*)(kbase + (size_t)key * 64 + d0);
    const uint2* gv0 = (const uint2*)(vbase + (size_t)key * 64 + d0);

    // prologue: stage tile 0 into buffer 0
    {
        uint2 rk[8], rv[8];
        #pragma unroll
        for (int i = 0; i < 8; ++i) rk[i] = gk0[i];
        #pragma unroll
        for (int i = 0; i < 8; ++i) rv[i] = gv0[i];
        #pragma unroll
        for (int i = 0; i < 8; ++i)
            *(uint2*)&Kl[0][key][d0 + i*4] = rk[i];
        #pragma unroll
        for (int i = 0; i < 8; ++i) {
            const uint2 u = rv[i];
            Vt[0][d0 + i*4 + 0][key] = (uint16_t)(u.x & 0xFFFF);
            Vt[0][d0 + i*4 + 1][key] = (uint16_t)(u.x >> 16);
            Vt[0][d0 + i*4 + 2][key] = (uint16_t)(u.y & 0xFFFF);
            Vt[0][d0 + i*4 + 3][key] = (uint16_t)(u.y >> 16);
        }
    }

    int cur = 0;
    for (int kt = 0; kt < 28; ++kt) {
        const bool has_next = (kt + 1) < 28;
        uint2 rk[8], rv[8];
        if (has_next) {
            const uint2* gk = gk0 + (size_t)(kt + 1) * 512;  // 64*64/8 uint2
            const uint2* gv = gv0 + (size_t)(kt + 1) * 512;
            #pragma unroll
            for (int i = 0; i < 8; ++i) rk[i] = gk[i];
            #pragma unroll
            for (int i = 0; i < 8; ++i) rv[i] = gv[i];
        }
        lds_barrier();   // buf[cur] visible; global loads stay in flight

        // S = Q K^T  (Q pre-scaled by 0.125 at pack time)
        v8f s[4] = {};
        #pragma unroll
        for (int nt = 0; nt < 4; ++nt) {
            #pragma unroll
            for (int kc = 0; kc < 2; ++kc) {
                Frag16 bfr;
                const uint16_t* rp = &Kl[cur][nt*16 + l15][kc*32 + lh*16];
                #pragma unroll
                for (int r = 0; r < 8; ++r)
                    bfr.d[r] = *(const uint32_t*)(rp + 2*r);
                s[nt] = __builtin_amdgcn_wmma_f32_16x16x32_bf16(
                    false, qf[kc].v, false, bfr.v, (short)0, s[nt], false, false);
            }
        }

        float mb[4];
        #pragma unroll
        for (int nt = 0; nt < 4; ++nt) {
            const int ky = kt*64 + nt*16 + l15;
            mb[nt] = (ky < 256 && mask[b*256 + ky] == 0) ? -1e30f : 0.0f;
        }
        #pragma unroll
        for (int nt = 0; nt < 4; ++nt)
            #pragma unroll
            for (int r = 0; r < 8; ++r)
                s[nt][r] = s[nt][r] + mb[nt];

        #pragma unroll
        for (int r = 0; r < 8; ++r) {
            float v = fmaxf(fmaxf(s[0][r], s[1][r]), fmaxf(s[2][r], s[3][r]));
            #pragma unroll
            for (int off = 1; off < 16; off <<= 1)
                v = fmaxf(v, __shfl_xor(v, off, 32));
            const float nm = fmaxf(mrow[r], v);
            const float sf = __expf(mrow[r] - nm);
            mrow[r] = nm;
            float ps = 0.0f;
            #pragma unroll
            for (int nt = 0; nt < 4; ++nt) {
                const float p = __expf(s[nt][r] - nm);
                s[nt][r] = p;
                ps += p;
            }
            #pragma unroll
            for (int off = 1; off < 16; off <<= 1)
                ps += __shfl_xor(ps, off, 32);
            lrow[r] = lrow[r] * sf + ps;
            #pragma unroll
            for (int dt = 0; dt < 4; ++dt)
                oacc[dt][r] *= sf;
        }

        #pragma unroll
        for (int nt = 0; nt < 4; ++nt)
            #pragma unroll
            for (int r = 0; r < 8; ++r)
                Pl[wv][lh*8 + r][nt*16 + l15] = f2bf(s[nt][r]);

        asm volatile("s_wait_dscnt 0" ::: "memory");   // wave-local P round-trip

        #pragma unroll
        for (int kc = 0; kc < 2; ++kc) {
            Frag16 pf;
            const uint16_t* rp = &Pl[wv][l15][kc*32 + lh*16];
            #pragma unroll
            for (int r = 0; r < 4; ++r) {
                pf.d[r]     = *(const uint32_t*)(rp + 2*r);
                pf.d[r + 4] = *(const uint32_t*)(rp + 8 + 2*r);
            }
            #pragma unroll
            for (int dt = 0; dt < 4; ++dt) {
                Frag16 vf;
                const uint16_t* vp = &Vt[cur][dt*16 + l15][kc*32 + lh*16];
                #pragma unroll
                for (int r = 0; r < 8; ++r)
                    vf.d[r] = *(const uint32_t*)(vp + 2*r);
                oacc[dt] = __builtin_amdgcn_wmma_f32_16x16x32_bf16(
                    false, pf.v, false, vf.v, (short)0, oacc[dt], false, false);
            }
        }

        if (has_next) {   // loadcnt wait lands here, after all the compute
            #pragma unroll
            for (int i = 0; i < 8; ++i)
                *(uint2*)&Kl[cur ^ 1][key][d0 + i*4] = rk[i];
            #pragma unroll
            for (int i = 0; i < 8; ++i) {
                const uint2 u = rv[i];
                Vt[cur ^ 1][d0 + i*4 + 0][key] = (uint16_t)(u.x & 0xFFFF);
                Vt[cur ^ 1][d0 + i*4 + 1][key] = (uint16_t)(u.x >> 16);
                Vt[cur ^ 1][d0 + i*4 + 2][key] = (uint16_t)(u.y & 0xFFFF);
                Vt[cur ^ 1][d0 + i*4 + 3][key] = (uint16_t)(u.y >> 16);
            }
        }
        cur ^= 1;
    }

    #pragma unroll
    for (int r = 0; r < 8; ++r) {
        const float inv = 1.0f / lrow[r];
        const int lq = q0 + lh*8 + r;
        #pragma unroll
        for (int dt = 0; dt < 4; ++dt) {
            const uint16_t v = f2bf(oacc[dt][r] * inv);
            const int col = h*64 + dt*16 + l15;
            if (lq < 256)
                out_ctx[((size_t)(b*256 + lq)) * 1024 + col] = v;
            else
                out_in[((size_t)(b*1536 + (lq - 256))) * 1024 + col] = v;
        }
    }
}

// ---------------------------------------------------------------------------
extern "C" void kernel_launch(void* const* d_in, const int* in_sizes, int n_in,
                              void* d_out, int out_size, void* d_ws, size_t ws_size,
                              hipStream_t stream) {
    const float* input    = (const float*)d_in[0];
    const float* context  = (const float*)d_in[1];
    const float* cosb     = (const float*)d_in[2];
    const float* sinb     = (const float*)d_in[3];
    const float* Wqkv_in  = (const float*)d_in[4];
    const float* bqkv_in  = (const float*)d_in[5];
    const float* Wqkv_ctx = (const float*)d_in[6];
    const float* bqkv_ctx = (const float*)d_in[7];
    const float* qw_in    = (const float*)d_in[8];
    const float* kw_in    = (const float*)d_in[9];
    const float* qw_ctx   = (const float*)d_in[10];
    const float* kw_ctx   = (const float*)d_in[11];
    const float* Wo_in    = (const float*)d_in[12];
    const float* bo_in    = (const float*)d_in[13];
    const float* Wo_ctx   = (const float*)d_in[14];
    const float* bo_ctx   = (const float*)d_in[15];
    const int*   attn_mask= (const int*)d_in[16];

    char* ws = (char*)d_ws;
    size_t off = 0;
    auto alloc = [&](size_t bytes) -> void* {
        void* p = (void*)(ws + off);
        off += (bytes + 255) & ~(size_t)255;
        return p;
    };

    uint16_t* xin_b      = (uint16_t*)alloc((size_t)3072 * 1024 * 2);
    uint16_t* xctx_b     = (uint16_t*)alloc((size_t)512  * 1024 * 2);
    uint16_t* Wt_qkv_in  = (uint16_t*)alloc((size_t)3072 * 1024 * 2);
    uint16_t* Wt_qkv_ctx = (uint16_t*)alloc((size_t)3072 * 1024 * 2);
    uint16_t* Wt_o_in    = (uint16_t*)alloc((size_t)1024 * 1024 * 2);
    uint16_t* Wt_o_ctx   = (uint16_t*)alloc((size_t)1024 * 1024 * 2);
    float*    qkv_in     = (float*)   alloc((size_t)3072 * 3072 * 4);
    float*    qkv_ctx    = (float*)   alloc((size_t)512  * 3072 * 4);
    uint16_t* qpk        = (uint16_t*)alloc((size_t)2*16*1792*64 * 2);
    uint16_t* kpk        = (uint16_t*)alloc((size_t)2*16*1792*64 * 2);
    uint16_t* vpk        = (uint16_t*)alloc((size_t)2*16*1792*64 * 2);
    uint16_t* a_in_b     = (uint16_t*)alloc((size_t)3072 * 1024 * 2);
    uint16_t* a_ctx_b    = (uint16_t*)alloc((size_t)512  * 1024 * 2);

    // 0) precision prep: activations -> bf16, weights -> bf16 transposed
    convert_bf16_kernel<<<3072, 256, 0, stream>>>(input,   xin_b,  3072 * 1024);
    convert_bf16_kernel<<<512,  256, 0, stream>>>(context, xctx_b,  512 * 1024);
    transpose_w_kernel<<<dim3(48, 16), 256, 0, stream>>>(Wqkv_in,  Wt_qkv_in,  1024, 3072);
    transpose_w_kernel<<<dim3(48, 16), 256, 0, stream>>>(Wqkv_ctx, Wt_qkv_ctx, 1024, 3072);
    transpose_w_kernel<<<dim3(16, 16), 256, 0, stream>>>(Wo_in,    Wt_o_in,    1024, 1024);
    transpose_w_kernel<<<dim3(16, 16), 256, 0, stream>>>(Wo_ctx,   Wt_o_ctx,   1024, 1024);

    // 1) QKV projections
    gemm_bf16_kernel<<<dim3(24, 24), 256, 0, stream>>>(xin_b,  Wt_qkv_in,  bqkv_in,  qkv_in,  3072, 3072, 1024);
    gemm_bf16_kernel<<<dim3(24, 4),  256, 0, stream>>>(xctx_b, Wt_qkv_ctx, bqkv_ctx, qkv_ctx,  512, 3072, 1024);

    // 2) RMSNorm + RoPE + pack (ctx rows at positions 0..255, input at 256..1791)
    pack_norm_rope_kernel<<<512,  256, 0, stream>>>(qkv_ctx, qw_ctx, kw_ctx, cosb, sinb, qpk, kpk, vpk,  256,   0);
    pack_norm_rope_kernel<<<3072, 256, 0, stream>>>(qkv_in,  qw_in,  kw_in,  cosb, sinb, qpk, kpk, vpk, 1536, 256);

    // 3) Attention: B*H*28 q-tiles
    attn_kernel<<<896, 128, 0, stream>>>(qpk, kpk, vpk, attn_mask, a_in_b, a_ctx_b);

    // 4) Output projections -> d_out = [input_output | context_output]
    float* out_f = (float*)d_out;
    gemm_bf16_kernel<<<dim3(8, 24), 256, 0, stream>>>(a_in_b,  Wt_o_in,  bo_in,  out_f,          3072, 1024, 1024);
    gemm_bf16_kernel<<<dim3(8, 4),  256, 0, stream>>>(a_ctx_b, Wt_o_ctx, bo_ctx, out_f + 3145728, 512, 1024, 1024);
}
